// EarthAttention3DRelative_81999515615772
// MI455X (gfx1250) — compile-verified, matrix operand order checked by reading, not run on hardware
//
#include <hip/hip_runtime.h>
#include <hip/hip_bf16.h>

typedef __attribute__((ext_vector_type(16))) __bf16 v16bf;
typedef __attribute__((ext_vector_type(8)))  float  v8f;

#define HEADS   6
#define N_TOK   144
#define DIM     192
#define HD      32
#define BATCH   1024
#define NW      512
#define QKV_COLS 576
#define NPAD    160   // 144 keys padded to 5*32 for the PV K-loop

static __device__ __forceinline__ __bf16 f2bf(float f) { return (__bf16)f; }

// Fragment loader: source row-major with lane already positioned at
// row*ld + (lane>>4)*8. Elements 0..7 <- K = hi*8+0..7, 8..15 <- K = hi*8+16..23,
// matching the 16-bit A (and B^T) VGPR layout for 16x16x32 WMMA.
static __device__ __forceinline__ v16bf ld_frag(const __bf16* p) {
  union { uint4 u[2]; v16bf v; } t;
  t.u[0] = *reinterpret_cast<const uint4*>(p);
  t.u[1] = *reinterpret_cast<const uint4*>(p + 16);
  return t.v;
}

static __device__ __forceinline__ v8f bf_wmma(v16bf a, v16bf b, v8f c) {
  return __builtin_amdgcn_wmma_f32_16x16x32_bf16(false, a, false, b, (short)0, c,
                                                 false, false);
}

// ---------------- conversion kernels ----------------
__global__ void k_conv_x(const float* __restrict__ x, __bf16* __restrict__ xb, long n2) {
  long i = (long)blockIdx.x * blockDim.x + threadIdx.x;
  if (i < n2) {
    float2 f = reinterpret_cast<const float2*>(x)[i];
    union { __bf16 h[2]; unsigned u; } t;
    t.h[0] = f2bf(f.x); t.h[1] = f2bf(f.y);
    reinterpret_cast<unsigned*>(xb)[i] = t.u;
  }
}

__global__ void k_conv_w(const float* __restrict__ wq, const float* __restrict__ wo,
                         __bf16* __restrict__ wqT, __bf16* __restrict__ woT) {
  int t = blockIdx.x * blockDim.x + threadIdx.x;
  if (t < DIM * QKV_COLS) {
    int kd = t / QKV_COLS, j = t - kd * QKV_COLS;
    wqT[j * DIM + kd] = f2bf(wq[t]);               // store W^T (cols become rows)
  } else {
    int t2 = t - DIM * QKV_COLS;
    if (t2 < DIM * DIM) {
      int kd = t2 / DIM, j = t2 - kd * DIM;
      woT[j * DIM + kd] = f2bf(wo[t2]);
    }
  }
}

__global__ void k_pad_v(__bf16* __restrict__ vT) {
  int t = blockIdx.x * blockDim.x + threadIdx.x;   // BATCH*HEADS*HD*16
  if (t < BATCH * HEADS * HD * 16) {
    int pad = t & 15; int row = t >> 4;
    vT[(size_t)row * NPAD + N_TOK + pad] = (__bf16)0.0f;
  }
}

// ------- QKV projection: x[147456,192] @ Wqkv -> q,k,vT (bf16) --------------
// One wave computes a 16(M) x 64(N) tile: shared A fragment, 4 accumulators.
__global__ void __launch_bounds__(256)
k_qkv(const __bf16* __restrict__ xb, const __bf16* __restrict__ wqT,
      const float* __restrict__ bqkv,
      __bf16* __restrict__ qw, __bf16* __restrict__ kw, __bf16* __restrict__ vT) {
  int lane = threadIdx.x & 31;
  int wv   = threadIdx.x >> 5;
  int tile = blockIdx.x * 8 + wv;                  // 9216 * 9 = 82944 wave-tiles
  int mt = tile / 9, jg = tile - mt * 9;           // jg: 64-wide column group
  int l15 = lane & 15, hi = lane >> 4;

  const __bf16* A  = xb  + (size_t)(mt * 16 + l15) * DIM + hi * 8;
  const __bf16* B0 = wqT + (size_t)(jg * 64 + l15) * DIM + hi * 8;

  v8f acc[4] = {{}, {}, {}, {}};
#pragma unroll
  for (int kt = 0; kt < 6; ++kt) {
    v16bf a = ld_frag(A + kt * 32);
#pragma unroll
    for (int u = 0; u < 4; ++u)
      acc[u] = bf_wmma(a, ld_frag(B0 + (size_t)u * 16 * DIM + kt * 32), acc[u]);
  }

#pragma unroll
  for (int u = 0; u < 4; ++u) {
    int j   = jg * 64 + u * 16 + l15;
    int sel = j / DIM;                             // uniform within the 64-group
    int jj  = j - sel * DIM;
    int h   = jj >> 5;
    int d   = jj & 31;
    float bias = bqkv[j];
#pragma unroll
    for (int r = 0; r < 8; ++r) {
      int rowg = mt * 16 + r + hi * 8;
      int b_ = rowg / N_TOK;
      int n  = rowg - b_ * N_TOK;
      float val = acc[u][r] + bias;
      size_t bh = (size_t)b_ * HEADS + h;
      if (sel == 0)      qw[(bh * N_TOK + n) * HD + d] = f2bf(val * 0.17677669529663689f);
      else if (sel == 1) kw[(bh * N_TOK + n) * HD + d] = f2bf(val);
      else               vT[(bh * HD + d) * NPAD + n]  = f2bf(val);  // V transposed
    }
  }
}

// ---------------- attention: one block per (b,h), 9 waves --------------------
__global__ void __launch_bounds__(288)
k_attn(const __bf16* __restrict__ qw, const __bf16* __restrict__ kw,
       const __bf16* __restrict__ vT, const float* __restrict__ mask,
       const float* __restrict__ btab, __bf16* __restrict__ ao) {
  __shared__ __align__(16) __bf16 Pls[N_TOK * NPAD];   // 46 KB

  int bh = blockIdx.x;                 // [0, 6144)
  int b  = bh / HEADS, h = bh - b * HEADS;
  int lane = threadIdx.x & 31;
  int wv   = threadIdx.x >> 5;         // M-tile 0..8
  int l15 = lane & 15, hi = lane >> 4;

  // ---- S = (q*scale) @ K^T : 9 tiles per wave, 1 WMMA each (K=hd=32)
  v16bf qa = ld_frag(qw + ((size_t)bh * N_TOK + wv * 16 + l15) * HD + hi * 8);
  v8f acc[9];
#pragma unroll
  for (int nt = 0; nt < 9; ++nt) {
    v16bf kb = ld_frag(kw + ((size_t)bh * N_TOK + nt * 16 + l15) * HD + hi * 8);
    v8f z = {};
    acc[nt] = bf_wmma(qa, kb, z);
  }

  // ---- relative-position bias + window mask
  const float* mrow = mask + (size_t)(b & (NW - 1)) * N_TOK * N_TOK;
  int rz[8], ry[8], rx[8], rn[8];
#pragma unroll
  for (int r = 0; r < 8; ++r) {
    int n = wv * 16 + r + hi * 8;
    rn[r] = n;
    rz[r] = n / 72; int rem = n - rz[r] * 72;
    ry[r] = rem / 12; rx[r] = rem - ry[r] * 12;
  }
#pragma unroll
  for (int nt = 0; nt < 9; ++nt) {
    int m = nt * 16 + l15;
    int zm = m / 72; int rem = m - zm * 72;
    int ym = rem / 12; int xm = rem - ym * 12;
#pragma unroll
    for (int r = 0; r < 8; ++r) {
      int idx = (rz[r] - zm + 1) * 11 + (ry[r] - ym + 5) + (rx[r] - xm + 11);
      acc[nt][r] += btab[idx * HEADS + h] + mrow[rn[r] * N_TOK + m];
    }
  }

  // ---- softmax: rows live in (vgpr r, 16-lane half); reduce within each 16-lane half
#pragma unroll
  for (int r = 0; r < 8; ++r) {
    float mx = acc[0][r];
    for (int nt = 1; nt < 9; ++nt) mx = fmaxf(mx, acc[nt][r]);
    for (int s = 1; s <= 8; s <<= 1) mx = fmaxf(mx, __shfl_xor(mx, s, 32));
    float e[9]; float sum = 0.f;
    for (int nt = 0; nt < 9; ++nt) { e[nt] = __expf(acc[nt][r] - mx); sum += e[nt]; }
    for (int s = 1; s <= 8; s <<= 1) sum += __shfl_xor(sum, s, 32);
    float inv = 1.0f / sum;
    int row = rn[r];
    for (int nt = 0; nt < 9; ++nt)
      Pls[row * NPAD + nt * 16 + l15] = f2bf(e[nt] * inv);
    Pls[row * NPAD + N_TOK + l15] = (__bf16)0.0f;       // zero the K-pad columns
  }
  __syncthreads();

  // ---- out = P @ V : M=144, N=32, K padded to 160 (5 k-steps, 2 n-tiles)
  v8f o0 = {}, o1 = {};
  const __bf16* Pb = Pls + (wv * 16 + l15) * NPAD + hi * 8;
  const __bf16* V0 = vT + ((size_t)bh * HD + l15)      * NPAD + hi * 8;
  const __bf16* V1 = vT + ((size_t)bh * HD + 16 + l15) * NPAD + hi * 8;
#pragma unroll
  for (int kt = 0; kt < 5; ++kt) {
    v16bf pa = ld_frag(Pb + kt * 32);
    o0 = bf_wmma(pa, ld_frag(V0 + kt * 32), o0);
    o1 = bf_wmma(pa, ld_frag(V1 + kt * 32), o1);
  }

#pragma unroll
  for (int r = 0; r < 8; ++r) {
    int n = wv * 16 + r + hi * 8;
    size_t base = ((size_t)b * N_TOK + n) * DIM + h * HD;
    ao[base + l15]      = f2bf(o0[r]);
    ao[base + 16 + l15] = f2bf(o1[r]);
  }
}

// ------- output projection: ao[147456,192] @ Wo + bo -> f32 out --------------
// One wave computes a 16(M) x 64(N) tile.
__global__ void __launch_bounds__(256)
k_out(const __bf16* __restrict__ ao, const __bf16* __restrict__ woT,
      const float* __restrict__ bo, float* __restrict__ out) {
  int lane = threadIdx.x & 31;
  int wv   = threadIdx.x >> 5;
  int tile = blockIdx.x * 8 + wv;                  // 9216 * 3 = 27648 wave-tiles
  int mt = tile / 3, jg = tile - mt * 3;
  int l15 = lane & 15, hi = lane >> 4;

  const __bf16* A  = ao  + (size_t)(mt * 16 + l15) * DIM + hi * 8;
  const __bf16* B0 = woT + (size_t)(jg * 64 + l15) * DIM + hi * 8;

  v8f acc[4] = {{}, {}, {}, {}};
#pragma unroll
  for (int kt = 0; kt < 6; ++kt) {
    v16bf a = ld_frag(A + kt * 32);
#pragma unroll
    for (int u = 0; u < 4; ++u)
      acc[u] = bf_wmma(a, ld_frag(B0 + (size_t)u * 16 * DIM + kt * 32), acc[u]);
  }

#pragma unroll
  for (int u = 0; u < 4; ++u) {
    int j = jg * 64 + u * 16 + l15;
    float bias = bo[j];
#pragma unroll
    for (int r = 0; r < 8; ++r) {
      int rowg = mt * 16 + r + hi * 8;
      out[(size_t)rowg * DIM + j] = acc[u][r] + bias;
    }
  }
}

extern "C" void kernel_launch(void* const* d_in, const int* in_sizes, int n_in,
                              void* d_out, int out_size, void* d_ws, size_t ws_size,
                              hipStream_t stream) {
  (void)in_sizes; (void)n_in; (void)out_size; (void)ws_size;
  const float* x    = (const float*)d_in[0];
  const float* mask = (const float*)d_in[1];
  const float* wq   = (const float*)d_in[2];
  const float* bq   = (const float*)d_in[3];
  const float* wo   = (const float*)d_in[4];
  const float* bo   = (const float*)d_in[5];
  const float* bt   = (const float*)d_in[6];

  const size_t ROWS = (size_t)BATCH * N_TOK;                  // 147456
  char* ws = (char*)d_ws;
  size_t off = 0;
  auto take = [&](size_t bytes) { size_t o = off; off = (off + bytes + 255) & ~(size_t)255; return o; };

  __bf16* xb   = (__bf16*)(ws + take(ROWS * DIM * 2));                    // 56.6 MB
  __bf16* wqT  = (__bf16*)(ws + take((size_t)QKV_COLS * DIM * 2));
  __bf16* woT  = (__bf16*)(ws + take((size_t)DIM * DIM * 2));
  __bf16* qw   = (__bf16*)(ws + take((size_t)BATCH * HEADS * N_TOK * HD * 2));
  __bf16* kw   = (__bf16*)(ws + take((size_t)BATCH * HEADS * N_TOK * HD * 2));
  __bf16* vT   = (__bf16*)(ws + take((size_t)BATCH * HEADS * HD * NPAD * 2));
  __bf16* aout = (__bf16*)(ws + take(ROWS * DIM * 2));

  long n2 = (long)(ROWS * DIM / 2);                           // 14,155,776
  k_conv_x<<<(unsigned)(n2 / 256), 256, 0, stream>>>(x, xb, n2);
  k_conv_w<<<576, 256, 0, stream>>>(wq, wo, wqT, woT);
  k_pad_v<<<(BATCH * HEADS * HD * 16) / 256, 256, 0, stream>>>(vT);

  // 9216 M-tiles * 9 column-groups = 82944 wave-tiles, 8 waves/block
  k_qkv<<<82944 / 8, 256, 0, stream>>>(xb, wqT, bq, qw, kw, vT);

  k_attn<<<BATCH * HEADS, 288, 0, stream>>>(qw, kw, vT, mask, bt, aout);

  // 9216 * 3 = 27648 wave-tiles
  k_out<<<27648 / 8, 256, 0, stream>>>(aout, woT, bo, (float*)d_out);
}
